// Model_19095424598065
// MI455X (gfx1250) — matrix operand order repaired and checked by reading, not compiled, not run on hardware
//
#include <hip/hip_runtime.h>

// ---------------- problem constants ----------------
#define BT      16384      // B*L graphs
#define NN      64         // nodes
#define FPN     8          // feats per node
#define H       128        // hidden
#define MODES   8
#define G       8          // graphs per workgroup
#define NTHREADS 256
#define LN_EPS  1e-5f

// ---------------- LDS layout (bytes) ----------------
#define OFF_BUFA   0                         // [G*64][128] bf16 row-major : 131072
#define OFF_BUFB   131072                    // [G][128 col][64 row] bf16 (transposed) : 131072
#define OFF_XF     262144                    // [G][MODES][128] bf16 : 16384 (also x stage)
#define OFF_OF     278528                    // [G][128 col][MODES] bf16 transposed : 16384 (also Wp/bp stage)
#define OFF_ADJ    294912                    // [64][64] bf16 : 8192
#define OFF_U8T    303104                    // [16][64] bf16 (rows 8..15 zero) : 2048
#define OFF_U8P    305152                    // [64][32] bf16 (K-padded U8, cols 8..31 zero) : 4096
#define SMEM_BYTES 309248

// ---------------- workspace layout (bf16 elements) ----------------
#define WS_W0T   0            // [MODES][H][H]  w0T[k][j][i] = w0[i][j][k]
#define WS_W2T   131072
#define WS_WST1  262144       // [H][H]  WT[j][i] = W[i][j]
#define WS_WNT1  278528
#define WS_WST3  294912
#define WS_WNT3  311296
#define WS_WOT   327680       // [16][H]  WoT[f][k] = Wo[k][f], rows f>=8 zero
#define WS_TOTAL 329728

typedef __attribute__((ext_vector_type(16))) __bf16 v16bf;
typedef __attribute__((ext_vector_type(2)))  __bf16 v2bf;
typedef __attribute__((ext_vector_type(8)))  float  v8f;

struct Frag { unsigned int u[8]; };
static_assert(sizeof(Frag) == sizeof(v16bf), "frag size");

#if defined(__has_builtin)
#  if __has_builtin(__builtin_amdgcn_cvt_pk_bf16_f32)
#    define HAVE_CVT_PK_BF16 1
#  endif
#endif

__device__ __forceinline__ unsigned int f2bf_pk(float lo, float hi){
#ifdef HAVE_CVT_PK_BF16
  v2bf r = __builtin_amdgcn_cvt_pk_bf16_f32(lo, hi);
  return __builtin_bit_cast(unsigned int, r);
#else
  v2bf v = { (__bf16)lo, (__bf16)hi };      // fptrunc RNE; native cvt if available
  return __builtin_bit_cast(unsigned int, v);
#endif
}
__device__ __forceinline__ unsigned short f2bf(float x){
  __bf16 b = (__bf16)x;
  return __builtin_bit_cast(unsigned short, b);
}
__device__ __forceinline__ float bf2f(unsigned short h){
  __bf16 b = __builtin_bit_cast(__bf16, h);
  return (float)b;
}
__device__ __forceinline__ v8f wmma_bf16(const Frag& a, const Frag& b, v8f c){
  return __builtin_amdgcn_wmma_f32_16x16x32_bf16(
      false, __builtin_bit_cast(v16bf, a),
      false, __builtin_bit_cast(v16bf, b),
      (short)0, c, false, false);
}

// ===================================================================
// Prep: convert / transpose weights to bf16 packed for B-fragments.
//   B-frag VGPR j needs a row pair (k,k+1) of B at one column -> store W^T
//   so the pair is one contiguous aligned dword.
// ===================================================================
__global__ void prep_weights(const float* __restrict__ w0,
                             const float* __restrict__ w2,
                             const float* __restrict__ Ws1,
                             const float* __restrict__ Wn1,
                             const float* __restrict__ Ws3,
                             const float* __restrict__ Wn3,
                             const float* __restrict__ Wo,
                             unsigned short* __restrict__ ws)
{
  int idx = blockIdx.x * NTHREADS + threadIdx.x;
  if (idx >= WS_TOTAL) return;
  if (idx < 131072){                       // w0T[k][j][i] = w0[i][j][k]
    int i = idx & 127, j = (idx >> 7) & 127, k = idx >> 14;
    ws[idx] = f2bf(w0[(i*H + j)*MODES + k]);
    return;
  }
  int r = idx - 131072;
  if (r < 131072){                         // w2T
    int i = r & 127, j = (r >> 7) & 127, k = r >> 14;
    ws[idx] = f2bf(w2[(i*H + j)*MODES + k]);
    return;
  }
  r -= 131072;
  if (r < 65536){                          // four 128x128 transposes
    int which = r >> 14;                   // 0..3 : Ws1, Wn1, Ws3, Wn3
    int local = r & 16383;
    int i = local & 127, j = local >> 7;   // dst[j][i] = src[i][j]
    const float* src = (which == 0) ? Ws1 : (which == 1) ? Wn1
                     : (which == 2) ? Ws3 : Wn3;
    ws[idx] = f2bf(src[i*H + j]);
    return;
  }
  r -= 65536;                              // WoT[f][k] padded to 16 rows
  int k = r & 127, f = r >> 7;
  ws[idx] = (f < FPN) ? f2bf(Wo[k*FPN + f]) : (unsigned short)0;
}

// ===================================================================
// Spectral layer:  h <- U8 @ ( per-mode mix( U8^T @ h ) )   (all WMMA)
// ===================================================================
__device__ __forceinline__ void spectral_stage(
    unsigned short* bufA, unsigned short* xfB, unsigned short* ofBt,
    const unsigned short* u8t, const unsigned short* u8pA,
    const unsigned int* wT32, int tid, int lane, int wid)
{
  const int l15 = lane & 15;
  const int hiA = (lane >> 4) << 3;   // +8  K-offset, A frag upper lanes
  const int hiB = (lane >> 4) << 4;   // +16 for upper lanes; unused, B frags use hiA (+8)

  // ---- xf = U8^T @ h  (per graph; wave w -> graph w), M=16 (rows 8..15 zero)
  {
    const int g = wid;
    Frag a[2];
    #pragma unroll
    for (int ks = 0; ks < 2; ++ks){
      const unsigned int* rp = (const unsigned int*)(u8t + l15*64);
      #pragma unroll
      for (int j = 0; j < 8; ++j){
        int kb = 32*ks + 2*(j & 3) + ((j >> 2) << 4) + hiA;
        a[ks].u[j] = rp[kb >> 1];
      }
    }
    #pragma unroll 1
    for (int nt = 0; nt < 8; ++nt){
      int n = nt*16 + l15;
      v8f c = {};
      #pragma unroll
      for (int ks = 0; ks < 2; ++ks){
        Frag b;
        #pragma unroll
        for (int j = 0; j < 8; ++j){
          int kr = 32*ks + 2*(j & 3) + ((j >> 2) << 4) + hiA;   // node row
          unsigned int lo = bufA[(g*64 + kr)*H + n];
          unsigned int hi = bufA[(g*64 + kr + 1)*H + n];
          b.u[j] = lo | (hi << 16);
        }
        c = wmma_bf16(a[ks], b, c);
      }
      if (lane < 16){                                 // rows 0..7 = modes
        #pragma unroll
        for (int v = 0; v < 8; ++v)
          xfB[(g*MODES + v)*H + n] = f2bf(c[v]);
      }
    }
  }
  __syncthreads();

  // ---- of[g,k,:] : per mode k, rows = graphs (M = G=8, padded to 16)
  //      store TRANSPOSED: ofBt[g][n][k]
  {
    const int k = wid;
    Frag a[4];
    #pragma unroll
    for (int ks = 0; ks < 4; ++ks){
      #pragma unroll
      for (int j = 0; j < 8; ++j){
        int kb = 32*ks + 2*(j & 3) + ((j >> 2) << 4) + hiA;
        a[ks].u[j] = (l15 < G)
          ? ((const unsigned int*)(xfB + (l15*MODES + k)*H))[kb >> 1] : 0u;
      }
    }
    const unsigned int* wk = wT32 + (k*H*H >> 1);     // mode-k W^T
    #pragma unroll 1
    for (int nt = 0; nt < 8; ++nt){
      int n = nt*16 + l15;
      v8f c = {};
      #pragma unroll
      for (int ks = 0; ks < 4; ++ks){
        Frag b;
        #pragma unroll
        for (int j = 0; j < 8; ++j){
          int kb = 32*ks + 2*(j & 3) + ((j >> 2) << 4) + hiA;
          b.u[j] = wk[(n*H + kb) >> 1];
        }
        c = wmma_bf16(a[ks], b, c);
      }
      if (lane < 16){
        #pragma unroll
        for (int v = 0; v < 8; ++v)                   // row v = graph v
          ofBt[(v*H + n)*MODES + k] = f2bf(c[v]);
      }
    }
  }
  __syncthreads();

  // ---- h = U8 @ of   (K = 8 zero-padded to 32: one WMMA per tile)
  {
    const int g = wid;
    #pragma unroll 1
    for (int mt = 0; mt < 4; ++mt){
      Frag a;
      const unsigned int* rp = (const unsigned int*)(u8pA + (mt*16 + l15)*32);
      #pragma unroll
      for (int j = 0; j < 8; ++j){
        int kb = 2*(j & 3) + ((j >> 2) << 4) + hiA;   // K-pad cols are zero
        a.u[j] = rp[kb >> 1];
      }
      #pragma unroll 1
      for (int nt = 0; nt < 8; ++nt){
        int n = nt*16 + l15;
        Frag b;
        #pragma unroll
        for (int j = 0; j < 8; ++j) b.u[j] = 0;       // K rows 8..31 zero
        if (lane < 16){
          const unsigned int* bp2 =
            (const unsigned int*)(ofBt + (g*H + n)*MODES);
          #pragma unroll
          for (int j = 0; j < 4; ++j) b.u[j] = bp2[j];
        }
        v8f c = {};
        c = wmma_bf16(a, b, c);
        #pragma unroll
        for (int v = 0; v < 8; ++v)
          bufA[(g*64 + mt*16 + hiA + v)*H + n] = f2bf(c[v]);
      }
    }
  }
  __syncthreads();
}

// ===================================================================
// Graph conv:  o = LN(h@Ws + adj@(h@Wn) + bs + bn); h <- relu(o)
//   bufB holds (h@Wn) TRANSPOSED per graph: bufB[g][col n][row k]
// ===================================================================
__device__ __forceinline__ void graphconv_stage(
    unsigned short* bufA, unsigned short* bufB, const unsigned short* adjL,
    const unsigned int* WsT32, const unsigned int* WnT32,
    const float* bs, const float* bn, const float* gam, const float* bet,
    int tid, int lane, int wid)
{
  const int l15 = lane & 15;
  const int hiA = (lane >> 4) << 3;
  const int hiB = (lane >> 4) << 4;
  const int g = wid;                       // wave owns one graph

  // ---- bufB = (h @ Wn)^T  (per graph, column-major store)
  #pragma unroll 1
  for (int mt = 0; mt < 4; ++mt){
    const int rowBase = g*64 + mt*16;
    Frag a[4];
    const unsigned int* rp = (const unsigned int*)(bufA + (rowBase + l15)*H);
    #pragma unroll
    for (int ks = 0; ks < 4; ++ks)
      #pragma unroll
      for (int j = 0; j < 8; ++j){
        int kb = 32*ks + 2*(j & 3) + ((j >> 2) << 4) + hiA;
        a[ks].u[j] = rp[kb >> 1];
      }
    #pragma unroll 1
    for (int nt = 0; nt < 8; ++nt){
      int n = nt*16 + l15;
      v8f c = {};
      #pragma unroll
      for (int ks = 0; ks < 4; ++ks){
        Frag b;
        #pragma unroll
        for (int j = 0; j < 8; ++j){
          int kb = 32*ks + 2*(j & 3) + ((j >> 2) << 4) + hiA;
          b.u[j] = WnT32[(n*H + kb) >> 1];
        }
        c = wmma_bf16(a[ks], b, c);
      }
      // transposed store: this lane's 8 rows are contiguous -> 4 dwords
      unsigned int* dst =
        (unsigned int*)(bufB + (g*H + n)*64 + mt*16 + hiA);
      #pragma unroll
      for (int v = 0; v < 8; v += 2)
        dst[v >> 1] = f2bf_pk(c[v], c[v+1]);
    }
  }
  __syncthreads();

  // ---- o = h@Ws + adj@(h@Wn) + bias ; LayerNorm ; relu -> bufA (in place)
  float bsn[8], gv[8], bv[8];
  #pragma unroll
  for (int nt = 0; nt < 8; ++nt){
    int n = nt*16 + l15;
    bsn[nt] = bs[n] + bn[n];
    gv[nt] = gam[n]; bv[nt] = bet[n];
  }
  #pragma unroll 1
  for (int mt = 0; mt < 4; ++mt){
    const int rowBase = g*64 + mt*16;
    Frag a[4];
    const unsigned int* rp = (const unsigned int*)(bufA + (rowBase + l15)*H);
    #pragma unroll
    for (int ks = 0; ks < 4; ++ks)
      #pragma unroll
      for (int j = 0; j < 8; ++j){
        int kb = 32*ks + 2*(j & 3) + ((j >> 2) << 4) + hiA;
        a[ks].u[j] = rp[kb >> 1];
      }
    Frag aj[2];
    const unsigned int* ap = (const unsigned int*)(adjL + (mt*16 + l15)*64);
    #pragma unroll
    for (int ks = 0; ks < 2; ++ks)
      #pragma unroll
      for (int j = 0; j < 8; ++j){
        int kb = 32*ks + 2*(j & 3) + ((j >> 2) << 4) + hiA;
        aj[ks].u[j] = ap[kb >> 1];
      }
    v8f acc[8];
    #pragma unroll
    for (int nt = 0; nt < 8; ++nt){
      int n = nt*16 + l15;
      v8f c;
      #pragma unroll
      for (int v = 0; v < 8; ++v) c[v] = bsn[nt];
      #pragma unroll
      for (int ks = 0; ks < 4; ++ks){
        Frag b;
        #pragma unroll
        for (int j = 0; j < 8; ++j){
          int kb = 32*ks + 2*(j & 3) + ((j >> 2) << 4) + hiA;
          b.u[j] = WsT32[(n*H + kb) >> 1];
        }
        c = wmma_bf16(a[ks], b, c);
      }
      // adj term: B from transposed bufB -> dword pair loads
      const unsigned int* bp2 = (const unsigned int*)(bufB + (g*H + n)*64);
      #pragma unroll
      for (int ks = 0; ks < 2; ++ks){
        Frag b;
        #pragma unroll
        for (int j = 0; j < 8; ++j){
          int kr = 32*ks + 2*(j & 3) + ((j >> 2) << 4) + hiA;
          b.u[j] = bp2[kr >> 1];
        }
        c = wmma_bf16(aj[ks], b, c);
      }
      acc[nt] = c;
    }
    // LayerNorm across H: rows live in VGPR index, cols in lanes+ntiles.
    float rs[8], rq[8];
    #pragma unroll
    for (int v = 0; v < 8; ++v){ rs[v] = 0.f; rq[v] = 0.f; }
    #pragma unroll
    for (int nt = 0; nt < 8; ++nt)
      #pragma unroll
      for (int v = 0; v < 8; ++v){
        float xx = acc[nt][v]; rs[v] += xx; rq[v] += xx*xx;
      }
    #pragma unroll
    for (int mask = 1; mask < 16; mask <<= 1)
      #pragma unroll
      for (int v = 0; v < 8; ++v){
        rs[v] += __shfl_xor(rs[v], mask, 32);
        rq[v] += __shfl_xor(rq[v], mask, 32);
      }
    float mu[8], rstd[8];
    #pragma unroll
    for (int v = 0; v < 8; ++v){
      mu[v] = rs[v] * (1.0f/H);
      float var = rq[v]*(1.0f/H) - mu[v]*mu[v];
      rstd[v] = rsqrtf(var + LN_EPS);
    }
    #pragma unroll
    for (int nt = 0; nt < 8; ++nt){
      int n = nt*16 + l15;
      #pragma unroll
      for (int v = 0; v < 8; ++v){
        float val = (acc[nt][v] - mu[v])*rstd[v]*gv[nt] + bv[nt];
        val = fmaxf(val, 0.f);
        bufA[(rowBase + hiA + v)*H + n] = f2bf(val);
      }
    }
  }
  __syncthreads();
}

// ===================================================================
// Fused network: one workgroup = 8 graphs, h resident in LDS.
// ===================================================================
__global__ __launch_bounds__(NTHREADS, 1)
void fused_gsn(const float* __restrict__ x,
               const float* __restrict__ adj,
               const float* __restrict__ U,
               const float* __restrict__ Wp, const float* __restrict__ bp,
               const unsigned short* __restrict__ ws,   // packed bf16 weights
               const float* __restrict__ bs1, const float* __restrict__ bn1,
               const float* __restrict__ g1,  const float* __restrict__ be1,
               const float* __restrict__ bs3, const float* __restrict__ bn3,
               const float* __restrict__ g3,  const float* __restrict__ be3,
               const float* __restrict__ bo,
               float* __restrict__ out)
{
  extern __shared__ char smem[];
  unsigned short* bufA = (unsigned short*)(smem + OFF_BUFA);
  unsigned short* bufB = (unsigned short*)(smem + OFF_BUFB);
  unsigned short* xfB  = (unsigned short*)(smem + OFF_XF);
  unsigned short* ofBt = (unsigned short*)(smem + OFF_OF);
  unsigned short* adjL = (unsigned short*)(smem + OFF_ADJ);
  unsigned short* u8t  = (unsigned short*)(smem + OFF_U8T);
  unsigned short* u8pA = (unsigned short*)(smem + OFF_U8P);

  const unsigned int* w0T  = (const unsigned int*)(ws + WS_W0T);
  const unsigned int* w2T  = (const unsigned int*)(ws + WS_W2T);
  const unsigned int* WsT1 = (const unsigned int*)(ws + WS_WST1);
  const unsigned int* WnT1 = (const unsigned int*)(ws + WS_WNT1);
  const unsigned int* WsT3 = (const unsigned int*)(ws + WS_WST3);
  const unsigned int* WnT3 = (const unsigned int*)(ws + WS_WNT3);
  const unsigned int* WoT  = (const unsigned int*)(ws + WS_WOT);

  const int tid  = threadIdx.x;
  const int lane = tid & 31;
  const int wid  = tid >> 5;
  const int l15  = lane & 15;
  const int hiA  = (lane >> 4) << 3;
  const int hiB  = (lane >> 4) << 4;
  const int g0   = blockIdx.x * G;

  // ---- static LDS: adj, U8^T (16 rows, 8..15 zero), U8 K-padded to 32
  for (int i = tid; i < 64*64; i += NTHREADS) adjL[i] = f2bf(adj[i]);
  for (int i = tid; i < 16*64; i += NTHREADS){
    int r = i >> 6, m = i & 63;
    u8t[i] = (r < MODES) ? f2bf(U[m*64 + r]) : (unsigned short)0;
  }
  for (int i = tid; i < 64*32; i += NTHREADS){
    int m = i >> 5, k = i & 31;
    u8pA[i] = (k < MODES) ? f2bf(U[m*64 + k]) : (unsigned short)0;
  }
  // ---- stage x (b128 coalesced) into xf region, Wp/bp into of region
  {
    const float4* xsrc = (const float4*)(x + (size_t)g0*(NN*FPN));
    float4* xdst = (float4*)xfB;              // 1024 float4
    for (int t = tid; t < (G*NN*FPN)/4; t += NTHREADS) xdst[t] = xsrc[t];
    const float4* wsrc = (const float4*)Wp;
    float4* wdst = (float4*)ofBt;             // 256 + 32 float4
    for (int t = tid; t < (FPN*H)/4; t += NTHREADS) wdst[t] = wsrc[t];
    const float4* bsrc = (const float4*)bp;
    for (int t = tid; t < H/4; t += NTHREADS) wdst[(FPN*H)/4 + t] = bsrc[t];
  }
  __syncthreads();

  // ---- input projection (K = 8: VALU, dword-packed)  h = x @ Wp + bp
  {
    const float* xs  = (const float*)xfB;
    const float* wpf = (const float*)ofBt;
    const float* bpf = wpf + FPN*H;
    unsigned int* outA = (unsigned int*)bufA;
    for (int t = tid; t < (G*NN*H)/2; t += NTHREADS){
      int e = t*2;
      int g = e >> 13; int rem = e & 8191; int r = rem >> 7; int c = rem & 127;
      float a0 = bpf[c], a1 = bpf[c+1];
      const float* xr = xs + g*(NN*FPN) + r*FPN;
      #pragma unroll
      for (int f = 0; f < FPN; ++f){
        float xv = xr[f];
        a0 += xv * wpf[f*H + c];
        a1 += xv * wpf[f*H + c + 1];
      }
      outA[t] = f2bf_pk(a0, a1);
    }
  }
  __syncthreads();

  spectral_stage (bufA, xfB, ofBt, u8t, u8pA, w0T, tid, lane, wid);
  graphconv_stage(bufA, bufB, adjL, WsT1, WnT1, bs1, bn1, g1, be1, tid, lane, wid);
  spectral_stage (bufA, xfB, ofBt, u8t, u8pA, w2T, tid, lane, wid);
  graphconv_stage(bufA, bufB, adjL, WsT3, WnT3, bs3, bn3, g3, be3, tid, lane, wid);

  // ---- output projection via WMMA: out = h @ Wo + bo (cols f<8 valid)
  {
    const int g = wid;
    Frag bW[4];
    #pragma unroll
    for (int ks = 0; ks < 4; ++ks)
      #pragma unroll
      for (int j = 0; j < 8; ++j){
        int kb = 32*ks + 2*(j & 3) + ((j >> 2) << 4) + hiA;
        bW[ks].u[j] = WoT[(l15*H + kb) >> 1];   // rows f>=8 are zero-padded
      }
    float bof = bo[l15 & 7];
    #pragma unroll 1
    for (int mt = 0; mt < 4; ++mt){
      Frag a[4];
      const unsigned int* rp =
        (const unsigned int*)(bufA + (g*64 + mt*16 + l15)*H);
      #pragma unroll
      for (int ks = 0; ks < 4; ++ks)
        #pragma unroll
        for (int j = 0; j < 8; ++j){
          int kb = 32*ks + 2*(j & 3) + ((j >> 2) << 4) + hiA;
          a[ks].u[j] = rp[kb >> 1];
        }
      v8f c;
      #pragma unroll
      for (int v = 0; v < 8; ++v) c[v] = bof;
      #pragma unroll
      for (int ks = 0; ks < 4; ++ks) c = wmma_bf16(a[ks], bW[ks], c);
      if (l15 < FPN){
        int f = l15;
        #pragma unroll
        for (int v = 0; v < 8; ++v){
          int r = mt*16 + hiA + v;
          out[(size_t)(g0 + g)*(NN*FPN) + r*FPN + f] = c[v];
        }
      }
    }
  }
}

// ===================================================================
extern "C" void kernel_launch(void* const* d_in, const int* in_sizes, int n_in,
                              void* d_out, int out_size, void* d_ws, size_t ws_size,
                              hipStream_t stream)
{
  (void)in_sizes; (void)n_in; (void)out_size; (void)ws_size;
  const float* x   = (const float*)d_in[0];
  const float* adj = (const float*)d_in[1];
  const float* U   = (const float*)d_in[2];
  const float* Wp  = (const float*)d_in[3];
  const float* bp  = (const float*)d_in[4];
  const float* w0  = (const float*)d_in[5];
  const float* w2  = (const float*)d_in[6];
  const float* Ws1 = (const float*)d_in[7];
  const float* bs1 = (const float*)d_in[8];
  const float* Wn1 = (const float*)d_in[9];
  const float* bn1 = (const float*)d_in[10];
  const float* g1  = (const float*)d_in[11];
  const float* be1 = (const float*)d_in[12];
  const float* Ws3 = (const float*)d_in[13];
  const float* bs3 = (const float*)d_in[14];
  const float* Wn3 = (const float*)d_in[15];
  const float* bn3 = (const float*)d_in[16];
  const float* g3  = (const float*)d_in[17];
  const float* be3 = (const float*)d_in[18];
  const float* Wo  = (const float*)d_in[19];
  const float* bo  = (const float*)d_in[20];

  unsigned short* ws = (unsigned short*)d_ws;
  float* out = (float*)d_out;

  // allow >64KB dynamic LDS (320KB/WGP on gfx1250); ignore error if redundant
  (void)hipFuncSetAttribute((const void*)fused_gsn,
                            hipFuncAttributeMaxDynamicSharedMemorySize,
                            SMEM_BYTES);

  // stage 1: pack weights to bf16, transposed for dword B-fragment loads
  prep_weights<<<(WS_TOTAL + NTHREADS - 1) / NTHREADS, NTHREADS, 0, stream>>>(
      w0, w2, Ws1, Wn1, Ws3, Wn3, Wo, ws);

  // stage 2: fused network, 8 graphs per workgroup
  fused_gsn<<<BT / G, NTHREADS, SMEM_BYTES, stream>>>(
      x, adj, U, Wp, bp, ws,
      bs1, bn1, g1, be1, bs3, bn3, g3, be3, bo, out);
}
